// CustomRNN_1855425872461
// MI455X (gfx1250) — matrix operand order repaired and checked by reading, not compiled, not run on hardware
//
#include <hip/hip_runtime.h>
#include <math.h>

typedef __attribute__((ext_vector_type(16))) __bf16 v16bf;
typedef __attribute__((ext_vector_type(8)))  __bf16 v8bf;
typedef __attribute__((ext_vector_type(8)))  float  v8f;

namespace {
constexpr int S_LEN = 2000;
constexpr int B_SZ  = 32;        // batch == WMMA M dimension (2 tiles of 16)
constexpr int D_IN  = 80;
constexpr int HID   = 512;
constexpr int OUTW  = 2 * HID;   // 1024 (fwd|bwd concat)
constexpr int NKC   = HID / 32;  // 16 recurrent k-chunks
constexpr int NKREG = 8;         // k-chunks resident in VGPRs (128 VGPRs/wave)
constexpr int NKLDS = 8;         // k-chunks resident in LDS (pre-swizzled, 256 KB)
constexpr int AST   = 520;       // A row stride (elems); 1040B = 65*16 -> conflict-free b128
constexpr int NT    = 512;       // 16 waves; wave w owns output columns [32w, 32w+32)

constexpr int ABUF_ELEMS = B_SZ * AST;               // 16640
constexpr int BW_ELEMS   = NKLDS * 2 * HID * 16;     // 131072 swizzled B fragments
constexpr size_t SHMEM_BYTES = (size_t)(ABUF_ELEMS + BW_ELEMS) * 2;  // 295,424 B < 320 KB

// ---- phase 1 geometry ----
constexpr int MROWS = B_SZ * S_LEN;   // 64000 flat rows of x / xp
constexpr int MT    = MROWS / 16;     // 4000 M tiles
constexpr int NTL   = HID / 16;       // 32 N tiles per direction
constexpr int TILES = 2 * MT * NTL;   // 256000 (dir, mt, ntile) wave-tiles
constexpr int P1_WPB = 8;             // waves per block (256 threads)
constexpr int P1_BLOCKS = TILES / P1_WPB;  // 32000
}

// ================= Phase 2: bidirectional recurrent scan ======================
// grid = 2 (fwd/bwd). h = tanh(xp(t) + h @ W_hh^T), xp read from (and h written
// back to) the same out slot. W_hh^T: 8 k-chunks in VGPRs + 8 k-chunks in LDS.
__global__ __launch_bounds__(NT, 1)
void birnn_scan(const float* __restrict__ h0,
                const float* __restrict__ Whh_f,
                const float* __restrict__ Whh_b,
                float* __restrict__ out)
{
  extern __shared__ __align__(16) __bf16 smem[];
  __bf16* abuf = smem;                 // [32][AST]: h state, bf16
  __bf16* bwl  = smem + ABUF_ELEMS;    // swizzled W_hh^T fragments, chunks 8..15

  const int rev = (int)blockIdx.x;
  const float* __restrict__ W_hh = rev ? Whh_b : Whh_f;
  const int nof = rev ? HID : 0;

  const int tid  = (int)threadIdx.x;
  const int wave = tid >> 5;
  const int lane = tid & 31;
  const int lm   = lane & 15;
  const int q    = lane >> 4;

  // ---- register-resident recurrent chunks 0..7 (128 VGPRs) ----
  v16bf Bh[2][NKREG];
  #pragma unroll
  for (int nt = 0; nt < 2; ++nt) {
    const int n = (wave * 2 + nt) * 16 + lm;
    const float* __restrict__ whr = W_hh + (size_t)n * HID;
    #pragma unroll
    for (int kc = 0; kc < NKREG; ++kc) {
      const int k0 = kc * 32 + q * 16;
      v16bf f;
      #pragma unroll
      for (int j = 0; j < 16; ++j) f[j] = (__bf16)whr[k0 + j];
      Bh[nt][kc] = f;
    }
  }

  // ---- LDS-resident chunks 8..15, pre-swizzled: bwl[c][q][n][j] ----
  for (int i = tid; i < BW_ELEMS; i += NT) {
    const int j  = i & 15;
    const int n  = (i >> 4) & (HID - 1);
    const int qq = (i >> 13) & 1;
    const int c  = i >> 14;                       // 0..7 -> kc = c + 8
    const int k  = (c + NKREG) * 32 + qq * 16 + j;
    bwl[i] = (__bf16)W_hh[(size_t)n * HID + k];
  }

  // ---- h state init ----
  for (int i = tid; i < B_SZ * HID; i += NT) {
    const int m = i >> 9, c = i & (HID - 1);
    abuf[m * AST + c] = (__bf16)h0[i];
  }
  __syncthreads();

  for (int step = 0; step < S_LEN; ++step) {
    const int t = rev ? (S_LEN - 1 - step) : step;

    v8f acc[2][2];
    #pragma unroll
    for (int mt = 0; mt < 2; ++mt)
      #pragma unroll
      for (int nt = 0; nt < 2; ++nt)
        #pragma unroll
        for (int j = 0; j < 8; ++j) acc[mt][nt][j] = 0.0f;

    // h @ W_hh^T : [32 x 512] * [512 x 32-slab], 4 accumulator chains
    #pragma unroll
    for (int kc = 0; kc < NKC; ++kc) {
      const int kb = kc * 32;
      v16bf a[2];
      #pragma unroll
      for (int mt = 0; mt < 2; ++mt) {
        const __bf16* rp = &abuf[(mt * 16 + lm) * AST + kb + q * 16];
        const v8bf lo = *(const v8bf*)rp;
        const v8bf hi = *(const v8bf*)(rp + 8);
        a[mt] = __builtin_shufflevector(lo, hi,
                                        0,1,2,3,4,5,6,7,8,9,10,11,12,13,14,15);
      }
      #pragma unroll
      for (int nt = 0; nt < 2; ++nt) {
        v16bf b;
        if (kc < NKREG) {
          b = Bh[nt][kc];
        } else {
          const int c = kc - NKREG;
          const __bf16* bp = &bwl[(size_t)c * 16384 + q * 8192 +
                                  ((wave * 2 + nt) * 16 + lm) * 16];
          const v8bf blo = *(const v8bf*)bp;
          const v8bf bhi = *(const v8bf*)(bp + 8);
          b = __builtin_shufflevector(blo, bhi,
                                      0,1,2,3,4,5,6,7,8,9,10,11,12,13,14,15);
        }
        #pragma unroll
        for (int mt = 0; mt < 2; ++mt)
          acc[mt][nt] = __builtin_amdgcn_wmma_f32_16x16x32_bf16(
              false, a[mt], false, b, (short)0, acc[mt][nt], false, false);
      }
    }

    // pre-activation xp (incl. both biases) from out; latency overlaps barrier
    float xv[2][2][8];
    #pragma unroll
    for (int nt = 0; nt < 2; ++nt) {
      const int n = (wave * 2 + nt) * 16 + lm;
      #pragma unroll
      for (int mt = 0; mt < 2; ++mt)
        #pragma unroll
        for (int r = 0; r < 8; ++r) {
          const int m = mt * 16 + q * 8 + r;
          xv[nt][mt][r] = out[((size_t)m * S_LEN + t) * OUTW + nof + n];
        }
    }

    __syncthreads();   // all LDS reads of step t done before h is overwritten

    #pragma unroll
    for (int nt = 0; nt < 2; ++nt) {
      const int n = (wave * 2 + nt) * 16 + lm;
      #pragma unroll
      for (int mt = 0; mt < 2; ++mt)
        #pragma unroll
        for (int r = 0; r < 8; ++r) {
          const int m = mt * 16 + q * 8 + r;
          const float v = tanhf(acc[mt][nt][r] + xv[nt][mt][r]);
          abuf[m * AST + n] = (__bf16)v;
          out[((size_t)m * S_LEN + t) * OUTW + nof + n] = v;   // overwrite xp with h
        }
    }

    // prefetch next step's xp (global_prefetch_b8)
    if (step + 1 < S_LEN) {
      const int tn = rev ? (S_LEN - 2 - step) : (step + 1);
      const int n0 = wave * 32 + lm;
      #pragma unroll
      for (int mt = 0; mt < 2; ++mt)
        __builtin_prefetch(&out[((size_t)(mt * 16 + q * 8) * S_LEN + tn) * OUTW + nof + n0],
                           0, 1);
    }
    __syncthreads();
  }
}

// ================= Phase 1: xp = x @ W_ih^T + b_ih + b_hh -> out ==============
// One 16x16 output tile per wave, K = 80 padded to 96 (3 bf16 WMMA chunks).
// Fully initializes out; the scan kernel consumes slot (m,t) then overwrites it.
__global__ __launch_bounds__(256, 1)
void xproj_wmma(const float* __restrict__ x,
                const float* __restrict__ Wih_f, const float* __restrict__ bih_f,
                const float* __restrict__ bhh_f,
                const float* __restrict__ Wih_b, const float* __restrict__ bih_b,
                const float* __restrict__ bhh_b,
                float* __restrict__ out)
{
  const int tid  = (int)threadIdx.x;
  const int wave = tid >> 5;
  const int lane = tid & 31;
  const int lm   = lane & 15;
  const int qh   = lane >> 4;

  const int tile = (int)blockIdx.x * P1_WPB + wave;
  const int dir  = tile / (MT * NTL);
  const int rem  = tile % (MT * NTL);
  const int mt   = rem / NTL;
  const int ntl  = rem % NTL;

  const float* __restrict__ W_ih = dir ? Wih_b : Wih_f;
  const float* __restrict__ b_ih = dir ? bih_b : bih_f;
  const float* __restrict__ b_hh = dir ? bhh_b : bhh_f;

  const int m = mt * 16 + lm;        // A row (both lane halves hold rows 0..15)
  const int n = ntl * 16 + lm;       // B column

  v8f acc;
  #pragma unroll
  for (int j = 0; j < 8; ++j) acc[j] = 0.0f;

  #pragma unroll
  for (int c = 0; c < 3; ++c) {
    // A 16x32 bf16: lane holds K run kb + 16*(l/16) + j
    v16bf a;
    #pragma unroll
    for (int j = 0; j < 16; ++j) {
      const int d = c * 32 + qh * 16 + j;
      float v = 0.0f;
      if (d < D_IN) v = x[(size_t)m * D_IN + d];
      a[j] = (__bf16)v;
    }
    // B 32x16 bf16: lane holds n = l%16, K run kb + 16*(l/16) + j
    const int kb = c * 32 + qh * 16;
    v16bf b;
    if (kb < D_IN) {
      #pragma unroll
      for (int j = 0; j < 16; ++j) b[j] = (__bf16)W_ih[(size_t)n * D_IN + kb + j];
    } else {
      #pragma unroll
      for (int j = 0; j < 16; ++j) b[j] = (__bf16)0.0f;
    }
    acc = __builtin_amdgcn_wmma_f32_16x16x32_bf16(false, a, false, b,
                                                  (short)0, acc, false, false);
  }

  const float bias = b_ih[n] + b_hh[n];
  #pragma unroll
  for (int r = 0; r < 8; ++r) {
    const int mm = mt * 16 + qh * 8 + r;     // C/D layout: M = 8*(lane/16) + vgpr
    out[(size_t)mm * OUTW + dir * HID + n] = acc[r] + bias;
  }
}

extern "C" void kernel_launch(void* const* d_in, const int* in_sizes, int n_in,
                              void* d_out, int out_size, void* d_ws, size_t ws_size,
                              hipStream_t stream) {
  (void)in_sizes; (void)n_in; (void)out_size; (void)d_ws; (void)ws_size;
  const float* x     = (const float*)d_in[0];
  const float* h0    = (const float*)d_in[1];
  const float* Wih_f = (const float*)d_in[2];
  const float* bih_f = (const float*)d_in[3];
  const float* Whh_f = (const float*)d_in[4];
  const float* bhh_f = (const float*)d_in[5];
  const float* Wih_b = (const float*)d_in[6];
  const float* bih_b = (const float*)d_in[7];
  const float* Whh_b = (const float*)d_in[8];
  const float* bhh_b = (const float*)d_in[9];
  float* out = (float*)d_out;

  hipLaunchKernelGGL(xproj_wmma, dim3(P1_BLOCKS), dim3(256), 0, stream,
                     x, Wih_f, bih_f, bhh_f, Wih_b, bih_b, bhh_b, out);
  hipLaunchKernelGGL(birnn_scan, dim3(2), dim3(NT), SHMEM_BYTES, stream,
                     h0, Whh_f, Whh_b, out);
}
